// SpatialTemporalPredictor_34540126994439
// MI455X (gfx1250) — compile-verified, matrix operand order checked by reading, not run on hardware
//
#include <hip/hip_runtime.h>
#include <hip/hip_bf16.h>

// ---------------------------------------------------------------------------
// SpatialTemporalPredictor forward for MI455X (gfx1250, wave32, WMMA + TDM).
// Dense GEMMs: v_wmma_f32_16x16x32_bf16, B tiles staged to LDS by the Tensor
// Data Mover (tensor_load_to_lds + s_wait_tensorcnt), A software-pipelined.
// Assumed d_in order (setup_inputs dict order, leaves flattened):
//  0:x 1:adj 2:ip_w 3:ip_b
//  4..7 : gat0 {W,a,g,b}   8..11: gat1 {W,a,g,b}
// 12..23: tl0 {in_w,in_b,out_w,out_b,w1,b1,w2,b2,n1g,n1b,n2g,n2b}
// 24..35: tl1 (same)
// 36:pw1 37:pb1 38:pw2 39:pb2
// ---------------------------------------------------------------------------

#define BB   8
#define TT   8
#define NN   512
#define FF   16
#define DD   128
#define HH   4
#define HD   32
#define ROWS (BB*TT*NN)   // 32768
#define SEQS (BB*NN)      // 4096

typedef __attribute__((ext_vector_type(16))) __bf16 v16bf;
typedef __attribute__((ext_vector_type(8)))  float  v8f;
typedef unsigned int u32x4 __attribute__((ext_vector_type(4)));
typedef int          i32x8 __attribute__((ext_vector_type(8)));
typedef int          i32x4 __attribute__((ext_vector_type(4)));

#if defined(__has_builtin)
#if __has_builtin(__builtin_amdgcn_tensor_load_to_lds) && \
    __has_builtin(__builtin_amdgcn_s_wait_tensorcnt)
#define HAVE_TDM 1
#endif
#endif

__device__ inline unsigned short f2bf(float f) {
  unsigned int u = __float_as_uint(f);
  unsigned int r = u + 0x7FFFu + ((u >> 16) & 1u);
  return (unsigned short)(r >> 16);
}

union FragU { uint4 q[2]; v16bf v; };

// 16-element bf16 WMMA fragment (ISA 7.12.2 16-bit A/B striping):
// lane holds k = kb_lane+{0..7} and kb_lane+16+{0..7} -> two aligned 16B loads
__device__ inline v16bf frag_ld(const unsigned short* __restrict__ p) {
  FragU u;
  u.q[0] = *(const uint4*)p;
  u.q[1] = *(const uint4*)(p + 16);
  return u.v;
}

// LDS byte offset of a __shared__ object: flat LDS addr low 32 bits (ISA 10.2)
__device__ inline unsigned lds_addr_of(const void* p) {
  return (unsigned)(unsigned long long)p;
}

#ifdef HAVE_TDM
// TDM 2D tile load: tile (til0 x til1) of 2-byte elements from a row-major
// tensor (dim0 x dim1, row stride stride0 elements) into LDS at lds_off.
// D# layout per CDNA5 ISA 8.3/8.4 (group0: count/lds/global/type; group1:
// data_size=2B, dims/tile/strides). Groups 2/3 zero (2D tensor); the
// clang-23 toolchain builtin takes an extra int32x8 (unused here) + cpol.
__device__ inline void tdm_load_2d_b16(unsigned lds_off, const void* gptr,
                                       unsigned dim0, unsigned dim1,
                                       unsigned til0, unsigned til1,
                                       unsigned stride0) {
  unsigned long long ga = (unsigned long long)gptr;
  u32x4 g0;
  g0[0] = 1u;                                   // count=1, user descriptor
  g0[1] = lds_off;                              // lds_addr
  g0[2] = (unsigned)ga;                         // global_addr[31:0]
  g0[3] = (unsigned)((ga >> 32) & 0x1FFFFFFu) | (2u << 30);  // [56:32] | type=2
  i32x8 g1;
  g1[0] = (int)(1u << 16);                      // workgroup_mask=0, data_size=1 (2B)
  g1[1] = (int)((dim0 & 0xFFFFu) << 16);        // tensor_dim0[15:0] -> bits 63:48
  g1[2] = (int)((dim0 >> 16) | ((dim1 & 0xFFFFu) << 16)); // dim0[31:16] | dim1[15:0]
  g1[3] = (int)((dim1 >> 16) | (til0 << 16));   // dim1[31:16] | tile_dim0
  g1[4] = (int)til1;                            // tile_dim1 (tile_dim2=0)
  g1[5] = (int)stride0;                         // tensor_dim0_stride[31:0]
  g1[6] = 0;                                    // stride hi / dim1_stride lo
  g1[7] = 0;
  i32x4 z4 = {0, 0, 0, 0};
  i32x8 z8 = {0, 0, 0, 0, 0, 0, 0, 0};
  __builtin_amdgcn_tensor_load_to_lds(g0, g1, z4, z4, z8, 0);
}
#endif

// ---------------------------------------------------------------------------
// bf16 GEMM:  C(MxN) = A(MxK,row-major) * W(NxK,row-major)^T + bias
// 128 threads = 4 waves; block tile 64x64 (wave tile 16x64). B tile staged in
// LDS once per block via TDM; A fragments software-pipelined from global.
// act: 0=none 1=relu 2=exact gelu.  Cf (fp32) and/or Cb (bf16) outputs.
// ---------------------------------------------------------------------------
__global__ __launch_bounds__(128) void gemm_bf16_kernel(
    const unsigned short* __restrict__ A, int lda,
    const unsigned short* __restrict__ Bw, int ldb,
    const float* __restrict__ bias,
    float* __restrict__ Cf, unsigned short* __restrict__ Cb, int ldc,
    int M, int N, int K, int act)
{
  __shared__ __align__(16) unsigned short s_b[64 * 512];   // up to K=512

  const int tid  = threadIdx.x;
  const int wave = tid >> 5;
  const int lane = tid & 31;
  const int n0 = blockIdx.x * 64;
  const int m0 = blockIdx.y * 64 + wave * 16;
  const int mf = lane & 15;
  const int kb_lane = (lane >> 4) << 3;

#ifdef HAVE_TDM
  if (wave == 0) {
    tdm_load_2d_b16(lds_addr_of(s_b), Bw + (size_t)n0 * ldb,
                    (unsigned)K, 64u, (unsigned)K, 64u, (unsigned)ldb);
    __builtin_amdgcn_s_wait_tensorcnt(0);
  }
  __syncthreads();
#else
  for (int r = wave; r < 64; r += 4) {
    const uint4* src = (const uint4*)(Bw + (size_t)(n0 + r) * ldb);
    uint4* dst = (uint4*)&s_b[r * K];
    for (int c = lane; c < (K >> 3); c += 32) dst[c] = src[c];
  }
  __syncthreads();
#endif

  const unsigned short* Arow = A + (size_t)(m0 + mf) * lda + kb_lane;
  v8f acc[4] = {};
  v16bf af = frag_ld(Arow);                 // prefetch k-step 0
  int kb = 0;
  for (; kb + 32 < K; kb += 32) {
    v16bf an = frag_ld(Arow + kb + 32);     // prefetch next step (hides VMEM)
#pragma unroll
    for (int j = 0; j < 4; ++j) {
      v16bf bf = frag_ld(&s_b[(j * 16 + mf) * K + kb + kb_lane]);  // ds_load
      acc[j] = __builtin_amdgcn_wmma_f32_16x16x32_bf16(
          false, af, false, bf, (short)0, acc[j], false, false);
    }
    af = an;
  }
#pragma unroll
  for (int j = 0; j < 4; ++j) {             // final k-step
    v16bf bf = frag_ld(&s_b[(j * 16 + mf) * K + kb + kb_lane]);
    acc[j] = __builtin_amdgcn_wmma_f32_16x16x32_bf16(
        false, af, false, bf, (short)0, acc[j], false, false);
  }

#pragma unroll
  for (int j = 0; j < 4; ++j) {
#pragma unroll
    for (int r = 0; r < 8; ++r) {
      int m = m0 + r + ((lane >> 4) << 3);
      int n = n0 + j * 16 + (lane & 15);
      float v = acc[j][r];
      if (bias) v += bias[n];
      if (act == 1)      v = fmaxf(v, 0.f);
      else if (act == 2) v = 0.5f * v * (1.f + erff(v * 0.70710678118654752f));
      size_t idx = (size_t)m * ldc + n;
      if (Cf) Cf[idx] = v;
      if (Cb) Cb[idx] = f2bf(v);
    }
  }
}

// f32 -> bf16 conversion
__global__ void f2bf_kernel(const float* __restrict__ in,
                            unsigned short* __restrict__ out, int n) {
  int t = blockIdx.x * 256 + threadIdx.x;
  if (t < n) out[t] = f2bf(in[t]);
}

// Input projection: h = x @ ip_w^T + ip_b  (K=16, memory bound -> scalar)
__global__ void inproj_kernel(const float* __restrict__ x,
                              const float* __restrict__ w,
                              const float* __restrict__ bia,
                              float* __restrict__ hf,
                              unsigned short* __restrict__ hb) {
  int t = blockIdx.x * 256 + threadIdx.x;   // ROWS*DD
  if (t >= ROWS * DD) return;
  int d = t & (DD - 1);
  int row = t >> 7;
  const float* xr = x + (size_t)row * FF;
  const float* wr = w + d * FF;
  float a = bia[d];
#pragma unroll
  for (int k = 0; k < FF; ++k) a += xr[k] * wr[k];
  hf[t] = a;
  hb[t] = f2bf(a);
}

// GAT src/dst head dots: src[row,h] = h[row, h*32: ] . a[h, :32], dst likewise
__global__ void srcdst_kernel(const float* __restrict__ hfull,
                              const float* __restrict__ aco,
                              float* __restrict__ srcv,
                              float* __restrict__ dstv) {
  int t = blockIdx.x * 256 + threadIdx.x;   // ROWS*HH
  if (t >= ROWS * HH) return;
  int h = t & 3;
  int row = t >> 2;
  const float* hr = hfull + (size_t)row * DD + h * HD;
  const float* ar = aco + h * 2 * HD;
  float s = 0.f, d2 = 0.f;
#pragma unroll
  for (int k = 0; k < HD; ++k) { s += hr[k] * ar[k]; d2 += hr[k] * ar[HD + k]; }
  srcv[t] = s;
  dstv[t] = d2;
}

// Build permuted+transposed value tensor vT[bt,h,d,j] = hh[bt, sj, sh, d]
// with idx = j*H+h, sh = idx/N, sj = idx%N (faithful torch-view replication).
__global__ void vperm_kernel(const unsigned short* __restrict__ hb,
                             unsigned short* __restrict__ vT) {
  int t = blockIdx.x * 256 + threadIdx.x;   // 64*4*32*512
  if (t >= BB * TT * HH * HD * NN) return;
  int j  = t & 511;
  int d  = (t >> 9) & 31;
  int h  = (t >> 14) & 3;
  int bt = t >> 16;
  int idx = j * HH + h;
  int sh = idx >> 9;        // idx / 512
  int sj = idx & 511;       // idx % 512
  vT[t] = hb[((size_t)bt * NN + sj) * DD + sh * HD + d];
}

// ---------------------------------------------------------------------------
// Fused GAT attention: per block (bt,h, 32 i-rows; 2 waves x 16 rows).
// TDM loads the (32x512) bf16 V tile into LDS asynchronously while waves
// compute masked leaky-relu scores + softmax numerators into LDS (bf16).
// Phase 2: (16x512) @ (512x32) via 32 WMMAs per wave, scaled by 1/sum.
// ---------------------------------------------------------------------------
__global__ __launch_bounds__(64) void gat_attn_kernel(
    const float* __restrict__ srcv, const float* __restrict__ dstv,
    const int* __restrict__ adj,
    const unsigned short* __restrict__ vT,
    float* __restrict__ hp)
{
  __shared__ __align__(16) unsigned short s_p[2][16][NN];   // 32 KB scores
  __shared__ __align__(16) unsigned short s_v[HD * NN];     // 32 KB V tile
  __shared__ float s_dst[NN];
  __shared__ float s_rinv[2][16];

  const int blk = blockIdx.x;
  const int ib = blk & 15;          // 16 i-blocks of 32 rows
  const int h  = (blk >> 4) & 3;
  const int bt = blk >> 6;
  const int tid = threadIdx.x;
  const int wave = tid >> 5;
  const int lane = tid & 31;
  const int rowbase = bt * NN;
  const unsigned short* vbase = vT + ((size_t)(bt * HH + h) * HD) * NN; // (32,512)

#ifdef HAVE_TDM
  if (wave == 0) {   // kick off async V-tile DMA; overlap with score phase
    tdm_load_2d_b16(lds_addr_of(s_v), vbase, NN, HD, NN, HD, NN);
  }
#else
  {
    const uint4* src = (const uint4*)vbase;
    uint4* dst = (uint4*)s_v;
    for (int c = tid; c < (HD * NN) / 8; c += 64) dst[c] = src[c];
  }
#endif

  for (int j = tid; j < NN; j += 64) s_dst[j] = dstv[(size_t)(rowbase + j) * HH + h];
  __syncthreads();

  const int i0 = ib * 32 + wave * 16;
  for (int il = 0; il < 16; ++il) {
    const int i = i0 + il;
    const float si = srcv[(size_t)(rowbase + i) * HH + h];
    const int* arow = adj + (size_t)i * NN;
    float mmax = -__builtin_inff();
    float evs[16];
#pragma unroll
    for (int c = 0; c < 16; ++c) {
      int j = lane + c * 32;
      float e = si + s_dst[j];
      e = (e > 0.f) ? e : 0.2f * e;                 // leaky_relu 0.2
      if (arow[j] == 0) e = -__builtin_inff();      // adjacency mask
      evs[c] = e;
      mmax = fmaxf(mmax, e);
    }
    for (int off = 16; off; off >>= 1) mmax = fmaxf(mmax, __shfl_xor(mmax, off, 32));
    float ssum = 0.f;
#pragma unroll
    for (int c = 0; c < 16; ++c) {
      int j = lane + c * 32;
      float p = (evs[c] == -__builtin_inff()) ? 0.f : __expf(evs[c] - mmax);
      ssum += p;
      s_p[wave][il][j] = f2bf(p);
    }
    for (int off = 16; off; off >>= 1) ssum += __shfl_xor(ssum, off, 32);
    if (lane == 0) s_rinv[wave][il] = 1.0f / ssum;
  }
#ifdef HAVE_TDM
  if (wave == 0) __builtin_amdgcn_s_wait_tensorcnt(0);
#endif
  __syncthreads();

  // Phase 2: WMMA over K=512, all operands in LDS
  v8f acc0 = {}, acc1 = {};
  const int mf = lane & 15;
  const int kb_lane = (lane >> 4) << 3;
  for (int kb = 0; kb < NN; kb += 32) {
    v16bf af = frag_ld(&s_p[wave][mf][kb + kb_lane]);
    v16bf b0 = frag_ld(&s_v[mf * NN + kb + kb_lane]);
    v16bf b1 = frag_ld(&s_v[(mf + 16) * NN + kb + kb_lane]);
    acc0 = __builtin_amdgcn_wmma_f32_16x16x32_bf16(false, af, false, b0, (short)0, acc0, false, false);
    acc1 = __builtin_amdgcn_wmma_f32_16x16x32_bf16(false, af, false, b1, (short)0, acc1, false, false);
  }
#pragma unroll
  for (int r = 0; r < 8; ++r) {
    int ml = r + ((lane >> 4) << 3);
    float sc = s_rinv[wave][ml];
    int gi = rowbase + i0 + ml;
    int n = lane & 15;
    hp[(size_t)gi * DD + h * HD + n]      = acc0[r] * sc;
    hp[(size_t)gi * DD + h * HD + 16 + n] = acc1[r] * sc;
  }
}

// Residual + LayerNorm (+optional relu); wave per row (D=128, 4 elems/lane)
__global__ __launch_bounds__(256) void ln_kernel(
    const float* __restrict__ X, const float* __restrict__ R,
    const float* __restrict__ g, const float* __restrict__ bta,
    float* __restrict__ Of, unsigned short* __restrict__ Ob,
    int rows, int relu)
{
  int wave = threadIdx.x >> 5, lane = threadIdx.x & 31;
  int row = blockIdx.x * 8 + wave;
  if (row >= rows) return;
  const float* x = X + (size_t)row * DD;
  const float* r = R ? R + (size_t)row * DD : nullptr;
  float v[4]; float s = 0.f, s2 = 0.f;
#pragma unroll
  for (int c = 0; c < 4; ++c) {
    float t = x[lane + c * 32];
    if (r) t += r[lane + c * 32];
    v[c] = t; s += t; s2 += t * t;
  }
  for (int off = 16; off; off >>= 1) { s += __shfl_xor(s, off, 32); s2 += __shfl_xor(s2, off, 32); }
  float mean = s * (1.f / DD);
  float var  = s2 * (1.f / DD) - mean * mean;
  float rstd = rsqrtf(var + 1e-5f);
#pragma unroll
  for (int c = 0; c < 4; ++c) {
    int d = lane + c * 32;
    float y = (v[c] - mean) * rstd * g[d] + bta[d];
    if (relu) y = fmaxf(y, 0.f);
    size_t idx = (size_t)row * DD + d;
    if (Of) Of[idx] = y;
    if (Ob) Ob[idx] = f2bf(y);
  }
}

// (B,T,N,D) -> (B,N,T,D) permute, emits fp32 + bf16
__global__ void permute_kernel(const float* __restrict__ in,
                               float* __restrict__ of,
                               unsigned short* __restrict__ ob) {
  int t = blockIdx.x * 256 + threadIdx.x;     // B*N*T*D = 4194304
  if (t >= BB * NN * TT * DD) return;
  int d  = t & 127;
  int tt = (t >> 7) & 7;
  int n  = (t >> 10) & 511;
  int b  = t >> 19;
  float v = in[(((size_t)b * TT + tt) * NN + n) * DD + d];
  of[t] = v;
  ob[t] = f2bf(v);
}

// Temporal attention (T=8, tiny) — scalar: one thread per (seq, head, tq)
__global__ __launch_bounds__(128) void tattn_kernel(
    const float* __restrict__ qkv, unsigned short* __restrict__ o)
{
  int t = blockIdx.x * 128 + threadIdx.x;     // SEQS*H*T = 131072
  if (t >= SEQS * HH * TT) return;
  int tq = t & 7;
  int h  = (t >> 3) & 3;
  int seq = t >> 5;
  const float* base = qkv + (size_t)seq * TT * (3 * DD);
  const float* q = base + tq * (3 * DD) + h * HD;
  float qv[HD];
#pragma unroll
  for (int d = 0; d < HD; ++d) qv[d] = q[d];
  float sc[TT]; float mmax = -__builtin_inff();
#pragma unroll
  for (int kt = 0; kt < TT; ++kt) {
    const float* k = base + kt * (3 * DD) + DD + h * HD;
    float dp = 0.f;
#pragma unroll
    for (int d = 0; d < HD; ++d) dp += qv[d] * k[d];
    sc[kt] = dp * 0.17677669529663687f;       // 1/sqrt(32)
    mmax = fmaxf(mmax, sc[kt]);
  }
  float ssum = 0.f;
#pragma unroll
  for (int kt = 0; kt < TT; ++kt) { sc[kt] = __expf(sc[kt] - mmax); ssum += sc[kt]; }
  float inv = 1.f / ssum;
  float acc[HD] = {};
#pragma unroll
  for (int kt = 0; kt < TT; ++kt) {
    const float* v = base + kt * (3 * DD) + 2 * DD + h * HD;
    float p = sc[kt] * inv;
#pragma unroll
    for (int d = 0; d < HD; ++d) acc[d] += p * v[d];
  }
  unsigned short* op = o + ((size_t)seq * TT + tq) * DD + h * HD;
#pragma unroll
  for (int d = 0; d < HD; ++d) op[d] = f2bf(acc[d]);
}

// Final head: out[row] = z[row,:] . pw2 + pb2
__global__ void head_kernel(const float* __restrict__ z,
                            const float* __restrict__ w,
                            const float* __restrict__ b2,
                            float* __restrict__ out) {
  int t = blockIdx.x * 256 + threadIdx.x;
  if (t >= SEQS) return;
  const float* zr = z + (size_t)t * DD;
  float a = b2[0];
#pragma unroll 8
  for (int k = 0; k < DD; ++k) a += zr[k] * w[k];
  out[t] = a;
}

// ---------------------------------------------------------------------------
extern "C" void kernel_launch(void* const* d_in, const int* in_sizes, int n_in,
                              void* d_out, int out_size, void* d_ws, size_t ws_size,
                              hipStream_t stream) {
  (void)in_sizes; (void)out_size;
  if (n_in < 40) return;

  // workspace layout (bytes)
  const size_t OFF_ACTA_F = 0;                          // 16 MB
  const size_t OFF_ACTA_B = OFF_ACTA_F + 16777216;      //  8 MB
  const size_t OFF_ACTB_F = OFF_ACTA_B + 8388608;       // 16 MB
  const size_t OFF_ACTB_B = OFF_ACTB_F + 16777216;      //  8 MB
  const size_t OFF_BIG    = OFF_ACTB_B + 8388608;       // 48 MB (qkv / vT / f1_bf)
  const size_t OFF_MID    = OFF_BIG + 50331648;         // 16 MB (hp / attn_out / f2)
  const size_t OFF_OBF    = OFF_MID + 16777216;         //  8 MB
  const size_t OFF_SRC    = OFF_OBF + 8388608;          // 0.5 MB
  const size_t OFF_DST    = OFF_SRC + 524288;           // 0.5 MB
  const size_t OFF_WBF    = OFF_DST + 524288;           // ~0.9 MB bf16 weights
  const size_t OFF_Z      = OFF_WBF + 884736;           //  2 MB
  const size_t NEED       = OFF_Z + 2097152;
  if (ws_size < NEED) return;

  char* ws = (char*)d_ws;
  float*          actA_f = (float*)(ws + OFF_ACTA_F);
  unsigned short* actA_b = (unsigned short*)(ws + OFF_ACTA_B);
  float*          actB_f = (float*)(ws + OFF_ACTB_F);
  unsigned short* actB_b = (unsigned short*)(ws + OFF_ACTB_B);
  float*          bigf   = (float*)(ws + OFF_BIG);
  unsigned short* bigb   = (unsigned short*)(ws + OFF_BIG);
  float*          midf   = (float*)(ws + OFF_MID);
  unsigned short* obf    = (unsigned short*)(ws + OFF_OBF);
  float*          srcv   = (float*)(ws + OFF_SRC);
  float*          dstv   = (float*)(ws + OFF_DST);
  unsigned short* wbf    = (unsigned short*)(ws + OFF_WBF);
  float*          zbuf   = (float*)(ws + OFF_Z);

  const float* x   = (const float*)d_in[0];
  const int*   adj = (const int*)d_in[1];
  const float* ipw = (const float*)d_in[2];
  const float* ipb = (const float*)d_in[3];
  const float* gW[2] = {(const float*)d_in[4], (const float*)d_in[8]};
  const float* gA[2] = {(const float*)d_in[5], (const float*)d_in[9]};
  const float* gG[2] = {(const float*)d_in[6], (const float*)d_in[10]};
  const float* gBv[2] = {(const float*)d_in[7], (const float*)d_in[11]};
  const int tlb[2] = {12, 24};
  const float* pw1 = (const float*)d_in[36];
  const float* pb1 = (const float*)d_in[37];
  const float* pw2 = (const float*)d_in[38];
  const float* pb2 = (const float*)d_in[39];
  float* out = (float*)d_out;

  // bf16 weight slots
  unsigned short* gWb[2] = {wbf, wbf + 16384};
  unsigned short* inwb[2], *outwb[2], *w1b[2], *w2b[2];
  size_t wo = 32768;
  for (int l = 0; l < 2; ++l) {
    inwb[l]  = wbf + wo; wo += 3 * DD * DD;       // 49152
    outwb[l] = wbf + wo; wo += DD * DD;           // 16384
    w1b[l]   = wbf + wo; wo += 4 * DD * DD;       // 65536
    w2b[l]   = wbf + wo; wo += 4 * DD * DD;       // 65536
  }
  unsigned short* pw1b = wbf + wo;                // 16384

  auto cvt = [&](const float* s, unsigned short* d, int n) {
    f2bf_kernel<<<(n + 255) / 256, 256, 0, stream>>>(s, d, n);
  };
  auto gemm = [&](const unsigned short* A, int lda, const unsigned short* Bw, int ldb,
                  const float* bias, float* Cf, unsigned short* Cb, int ldc,
                  int M, int Nn, int K, int act) {
    dim3 g(Nn / 64, M / 64);
    gemm_bf16_kernel<<<g, 128, 0, stream>>>(A, lda, Bw, ldb, bias, Cf, Cb, ldc, M, Nn, K, act);
  };

  // 0) weights -> bf16
  for (int l = 0; l < 2; ++l) {
    cvt(gW[l], gWb[l], DD * DD);
    cvt((const float*)d_in[tlb[l] + 0], inwb[l],  3 * DD * DD);
    cvt((const float*)d_in[tlb[l] + 2], outwb[l], DD * DD);
    cvt((const float*)d_in[tlb[l] + 4], w1b[l],   4 * DD * DD);
    cvt((const float*)d_in[tlb[l] + 6], w2b[l],   4 * DD * DD);
  }
  cvt(pw1, pw1b, DD * DD);

  // 1) input projection -> actA (fp32 + bf16)
  inproj_kernel<<<(ROWS * DD + 255) / 256, 256, 0, stream>>>(x, ipw, ipb, actA_f, actA_b);

  // 2) two GAT layers (batched over all 64 (b,t) graphs)
  for (int l = 0; l < 2; ++l) {
    // h = x @ W^T -> actB
    gemm(actA_b, DD, gWb[l], DD, nullptr, actB_f, actB_b, DD, ROWS, DD, DD, 0);
    // per-head src/dst scalars
    srcdst_kernel<<<(ROWS * HH + 255) / 256, 256, 0, stream>>>(actB_f, gA[l], srcv, dstv);
    // permuted/transposed V (bf16) into big
    vperm_kernel<<<(BB * TT * HH * HD * NN + 255) / 256, 256, 0, stream>>>(actB_b, bigb);
    // fused masked softmax + attn@V (WMMA, TDM-staged V) -> mid (hp)
    gat_attn_kernel<<<BB * TT * HH * (NN / 32), 64, 0, stream>>>(srcv, dstv, adj, bigb, midf);
    // relu(LN(hp + h)) -> actA
    ln_kernel<<<ROWS / 8, 256, 0, stream>>>(midf, actB_f, gG[l], gBv[l], actA_f, actA_b, ROWS, 1);
  }

  // 3) (B,T,N,D) -> (B,N,T,D) per-node sequences -> actB
  permute_kernel<<<(ROWS * DD + 255) / 256, 256, 0, stream>>>(actA_f, actB_f, actB_b);

  // 4) two transformer layers; xt lives in actB
  for (int l = 0; l < 2; ++l) {
    const float* inb  = (const float*)d_in[tlb[l] + 1];
    const float* outb = (const float*)d_in[tlb[l] + 3];
    const float* b1   = (const float*)d_in[tlb[l] + 5];
    const float* b2   = (const float*)d_in[tlb[l] + 7];
    const float* n1g  = (const float*)d_in[tlb[l] + 8];
    const float* n1b  = (const float*)d_in[tlb[l] + 9];
    const float* n2g  = (const float*)d_in[tlb[l] + 10];
    const float* n2b  = (const float*)d_in[tlb[l] + 11];

    // qkv = xt @ in_w^T + in_b -> big (fp32, ldc=384)
    gemm(actB_b, DD, inwb[l], DD, inb, bigf, nullptr, 3 * DD, ROWS, 3 * DD, DD, 0);
    // temporal attention (T=8, scalar) -> o (bf16)
    tattn_kernel<<<(SEQS * HH * TT + 127) / 128, 128, 0, stream>>>(bigf, obf);
    // out proj -> mid
    gemm(obf, DD, outwb[l], DD, outb, midf, nullptr, DD, ROWS, DD, DD, 0);
    // x1 = LN(xt + o) -> actA
    ln_kernel<<<ROWS / 8, 256, 0, stream>>>(midf, actB_f, n1g, n1b, actA_f, actA_b, ROWS, 0);
    // f1 = gelu(x1 @ w1^T + b1) -> big (bf16 only, ldc=512); qkv is dead
    gemm(actA_b, DD, w1b[l], DD, b1, nullptr, bigb, 4 * DD, ROWS, 4 * DD, DD, 2);
    // f2 = f1 @ w2^T + b2 -> mid
    gemm(bigb, 4 * DD, w2b[l], 4 * DD, b2, midf, nullptr, DD, ROWS, DD, 4 * DD, 0);
    // xt = LN(x1 + f2) -> actB
    ln_kernel<<<ROWS / 8, 256, 0, stream>>>(midf, actA_f, n2g, n2b, actB_f, actB_b, ROWS, 0);
  }

  // 5) head on last timestep: strided rows t=7 (lda = T*D)
  gemm(actB_b + (TT - 1) * DD, TT * DD, pw1b, DD, pb1, zbuf, nullptr, DD, SEQS, DD, DD, 1);
  head_kernel<<<(SEQS + 255) / 256, 256, 0, stream>>>(zbuf, pw2, pb2, out);
}
